// PartialRadialLayer_69724499084082
// MI455X (gfx1250) — matrix operand order repaired and checked by reading, not compiled, not run on hardware
//
#include <hip/hip_runtime.h>
#include <hip/hip_bf16.h>
#include <math.h>

typedef __attribute__((ext_vector_type(16))) _Float16 v16h;
typedef __attribute__((ext_vector_type(8)))  float    v8f;
typedef __attribute__((ext_vector_type(4)))  int      v4i_;

#define BATCH        65536
#define IN_W         64
#define INNER_W      32
#define N_LEAVES     256
#define N_NODES      255
#define DEPTH        8

// padded f16 T layout: [leaf][col n (32)][i (64)], row stride 72 halves = 144 B
// (16B-aligned rows; 36-dword stride -> 16 lanes hit disjoint bank quads)
#define T_COL_STRIDE 72
#define T_LEAF_H     (32 * T_COL_STRIDE)   // 2304 halves = 4608 B per leaf
#define T_TOTAL_H    (N_LEAVES * T_LEAF_H) // 1,179,648 B

// main-kernel tiling: 2 waves/block, 32 rows per wave
#define BLK_THREADS  64
#define BLK_ROWS     64
#define DIST_STRIDE  260                   // 256 + pad -> conflict-free scalar reads
#define GROUP_H      (2 * T_LEAF_H)        // 2 leaves per LDS group = 4608 halves
#define N_GROUPS     (N_LEAVES / 2)        // 128

#if defined(__has_builtin)
#if __has_builtin(__builtin_amdgcn_global_load_async_to_lds_b128)
#define HAVE_ASYNC_LDS 1
#endif
#endif

__device__ __forceinline__ float sigmoidf_(float z) {
    return 1.0f / (1.0f + __expf(-z));
}

__device__ __forceinline__ void wait_async0() {
#if defined(__has_builtin)
#if __has_builtin(__builtin_amdgcn_s_wait_asynccnt)
    __builtin_amdgcn_s_wait_asynccnt(0);
    return;
#else
    asm volatile("s_wait_asynccnt 0" ::: "memory");
    return;
#endif
#endif
}

// ---------------------------------------------------------------------------
// Prep 1: normalized ray + per-node (alpha, beta): decision = sig(alpha*t - beta)
// ---------------------------------------------------------------------------
__global__ void prep_misc(const float* __restrict__ ray,
                          const float* __restrict__ w_i,
                          const float* __restrict__ b_i,
                          const float* __restrict__ a_i,
                          float* __restrict__ rayN,
                          float* __restrict__ alphaW,
                          float* __restrict__ betaW) {
    __shared__ float rn;
    int t = threadIdx.x;
    if (t == 0) {
        float acc = 0.0f;
        for (int i = 0; i < IN_W; ++i) acc += ray[i] * ray[i];
        rn = fmaxf(sqrtf(acc), 1e-8f);
    }
    __syncthreads();
    if (t < IN_W) rayN[t] = ray[t] / rn;
    if (t < N_NODES) {
        float ap1 = 1.0f + a_i[t];
        alphaW[t] = (0.5f + sigmoidf_(w_i[t])) * ap1;
        betaW[t]  = sigmoidf_(b_i[t]) * ap1;
    }
}

// ---------------------------------------------------------------------------
// Prep 2: T (f32, [l][i][w]) -> f16 transposed+padded [l][w][i], stride 72
// ---------------------------------------------------------------------------
__global__ void prep_T(const float* __restrict__ T, _Float16* __restrict__ dst) {
    int idx = blockIdx.x * blockDim.x + threadIdx.x;
    if (idx >= N_LEAVES * IN_W * INNER_W) return;
    int i = idx & 63;
    int w = (idx >> 6) & 31;
    int l = idx >> 11;
    float v = T[l * (IN_W * INNER_W) + i * INNER_W + w];
    dst[l * T_LEAF_H + w * T_COL_STRIDE + i] = (_Float16)v;
}

// ---------------------------------------------------------------------------
// Main kernel
// ---------------------------------------------------------------------------
__device__ __forceinline__ void loadA(const float* xp, v16h& xa, v16h& xb) {
    #pragma unroll
    for (int c = 0; c < 2; ++c) {
        const float4 f0 = *(const float4*)(xp + c * 8);
        const float4 f1 = *(const float4*)(xp + c * 8 + 4);
        const float4 f2 = *(const float4*)(xp + 32 + c * 8);
        const float4 f3 = *(const float4*)(xp + 32 + c * 8 + 4);
        xa[c * 8 + 0] = (_Float16)f0.x; xa[c * 8 + 1] = (_Float16)f0.y;
        xa[c * 8 + 2] = (_Float16)f0.z; xa[c * 8 + 3] = (_Float16)f0.w;
        xa[c * 8 + 4] = (_Float16)f1.x; xa[c * 8 + 5] = (_Float16)f1.y;
        xa[c * 8 + 6] = (_Float16)f1.z; xa[c * 8 + 7] = (_Float16)f1.w;
        xb[c * 8 + 0] = (_Float16)f2.x; xb[c * 8 + 1] = (_Float16)f2.y;
        xb[c * 8 + 2] = (_Float16)f2.z; xb[c * 8 + 3] = (_Float16)f2.w;
        xb[c * 8 + 4] = (_Float16)f3.x; xb[c * 8 + 5] = (_Float16)f3.y;
        xb[c * 8 + 6] = (_Float16)f3.z; xb[c * 8 + 7] = (_Float16)f3.w;
    }
}

__device__ __forceinline__ v16h splat16(_Float16 s) {
    v16h v;
    #pragma unroll
    for (int i = 0; i < 16; ++i) v[i] = s;
    return v;
}

__launch_bounds__(BLK_THREADS)
__global__ void radial_tree_wmma(const float* __restrict__ x,
                                 const _Float16* __restrict__ Tt,
                                 const float* __restrict__ rayN,
                                 const float* __restrict__ alphaW,
                                 const float* __restrict__ betaW,
                                 float* __restrict__ out) {
    __shared__ _Float16 distS[BLK_ROWS * DIST_STRIDE];        // 33,280 B
    __shared__ float    angleS[BLK_ROWS];                      //    256 B
    __shared__ __align__(16) _Float16 tbuf[2 * GROUP_H];       // 18,432 B

    const int tid     = threadIdx.x;
    const int rowBase = blockIdx.x * BLK_ROWS;

    // ---- Phase 1: per-row angle = acos(clip(cos)) / pi -------------------
    {
        const float4* xr = (const float4*)(x + (size_t)(rowBase + tid) * IN_W);
        const float4* ur = (const float4*)rayN;
        float dot = 0.0f, ss = 0.0f;
        #pragma unroll
        for (int j = 0; j < IN_W / 4; ++j) {
            float4 v = xr[j];
            float4 u = ur[j];
            dot += v.x * u.x + v.y * u.y + v.z * u.z + v.w * u.w;
            ss  += v.x * v.x + v.y * v.y + v.z * v.z + v.w * v.w;
        }
        float xn = fmaxf(sqrtf(ss), 1e-8f);
        float c  = fminf(1.0f, fmaxf(-1.0f, dot / xn));
        angleS[tid] = acosf(c) * 0.31830988618379067f;   // 1/pi
    }
    __syncthreads();

    // ---- Phase 2: in-place tree expansion of leaf distribution -----------
    for (int d = 0; d < DEPTH; ++d) {
        int tasks = BLK_ROWS << d;
        for (int task = tid; task < tasks; task += BLK_THREADS) {
            int j    = task & ((1 << d) - 1);
            int row  = task >> d;
            int node = (1 << d) - 1 + j;
            float g  = sigmoidf_(alphaW[node] * angleS[row] - betaW[node]);
            int S2   = N_LEAVES >> d;
            int half = S2 >> 1;
            int base = row * DIST_STRIDE + j * S2;
            float v  = d ? (float)distS[base] : 1.0f;
            distS[base]        = (_Float16)(v * g);
            distS[base + half] = (_Float16)(v * (1.0f - g));
        }
        __syncthreads();
    }

    // ---- Phase 3: x row slices -> A registers (two row-tiles per wave) ----
    const int wv   = tid >> 5;        // wave 0..1
    const int lane = tid & 31;
    const int lm   = lane & 15;       // A-row-in-tile / B-column
    const int kh   = lane >> 4;       // K-half select
    const int r0   = wv * 32 + lm;    // local row, tile 0
    const int r1   = r0 + 16;         // local row, tile 1

    v16h xa0, xb0, xa1, xb1;
    loadA(x + (size_t)(rowBase + r0) * IN_W + kh * 16, xa0, xb0);
    loadA(x + (size_t)(rowBase + r1) * IN_W + kh * 16, xa1, xb1);

    // ---- Phase 4: double-buffered B staging + WMMA K-loop ----------------
    v8f acc00 = {0.f,0.f,0.f,0.f,0.f,0.f,0.f,0.f};   // rows tile0, cols 0..15
    v8f acc01 = acc00;                               // rows tile0, cols 16..31
    v8f acc10 = acc00;                               // rows tile1, cols 0..15
    v8f acc11 = acc00;                               // rows tile1, cols 16..31

    auto stage = [&](int g, int b) {
        const uint4* src = (const uint4*)(Tt + (size_t)g * GROUP_H);
        uint4*       dst = (uint4*)(tbuf + b * GROUP_H);
        const int n16 = GROUP_H * 2 / 16;   // 576 chunks of 16 B
#if HAVE_ASYNC_LDS
        for (int k = tid; k < n16; k += BLK_THREADS) {
            __builtin_amdgcn_global_load_async_to_lds_b128(
                (__attribute__((address_space(1))) v4i_*)(src + k),
                (__attribute__((address_space(3))) v4i_*)(dst + k),
                0, 0);
        }
#else
        for (int k = tid; k < n16; k += BLK_THREADS) dst[k] = src[k];
#endif
    };

    stage(0, 0);
#if HAVE_ASYNC_LDS
    wait_async0();
#endif
    __syncthreads();

    const _Float16* distRow0 = distS + r0 * DIST_STRIDE;
    const _Float16* distRow1 = distS + r1 * DIST_STRIDE;

    for (int g = 0; g < N_GROUPS; ++g) {
        if (g + 1 < N_GROUPS) stage(g + 1, (g + 1) & 1);
        const _Float16* cur = tbuf + (g & 1) * GROUP_H;

        #pragma unroll
        for (int lg = 0; lg < 2; ++lg) {
            const _Float16* bl = cur + lg * T_LEAF_H;
            const v16h b00 = *(const v16h*)(bl + lm * T_COL_STRIDE + kh * 16);
            const v16h b10 = *(const v16h*)(bl + lm * T_COL_STRIDE + 32 + kh * 16);
            const v16h b01 = *(const v16h*)(bl + (16 + lm) * T_COL_STRIDE + kh * 16);
            const v16h b11 = *(const v16h*)(bl + (16 + lm) * T_COL_STRIDE + 32 + kh * 16);

            const v16h s0 = splat16(distRow0[g * 2 + lg]);
            const v16h s1 = splat16(distRow1[g * 2 + lg]);
            const v16h a00 = xa0 * s0, a01 = xb0 * s0;
            const v16h a10 = xa1 * s1, a11 = xb1 * s1;

            acc00 = __builtin_amdgcn_wmma_f32_16x16x32_f16(false, a00, false, b00,
                                                           (short)0, acc00, false, false);
            acc10 = __builtin_amdgcn_wmma_f32_16x16x32_f16(false, a10, false, b00,
                                                           (short)0, acc10, false, false);
            acc00 = __builtin_amdgcn_wmma_f32_16x16x32_f16(false, a01, false, b10,
                                                           (short)0, acc00, false, false);
            acc10 = __builtin_amdgcn_wmma_f32_16x16x32_f16(false, a11, false, b10,
                                                           (short)0, acc10, false, false);
            acc01 = __builtin_amdgcn_wmma_f32_16x16x32_f16(false, a00, false, b01,
                                                           (short)0, acc01, false, false);
            acc11 = __builtin_amdgcn_wmma_f32_16x16x32_f16(false, a10, false, b01,
                                                           (short)0, acc11, false, false);
            acc01 = __builtin_amdgcn_wmma_f32_16x16x32_f16(false, a01, false, b11,
                                                           (short)0, acc01, false, false);
            acc11 = __builtin_amdgcn_wmma_f32_16x16x32_f16(false, a11, false, b11,
                                                           (short)0, acc11, false, false);
        }
#if HAVE_ASYNC_LDS
        wait_async0();
#endif
        __syncthreads();
    }

    // ---- Phase 5: write out (C/D layout: VGPR v -> M = v + 8*kh) ---------
    const int m0 = rowBase + wv * 32 + kh * 8;       // tile0 rows
    #pragma unroll
    for (int v = 0; v < 8; ++v) {
        out[(size_t)(m0 + v) * INNER_W + lm]           = acc00[v];
        out[(size_t)(m0 + v) * INNER_W + 16 + lm]      = acc01[v];
        out[(size_t)(m0 + 16 + v) * INNER_W + lm]      = acc10[v];
        out[(size_t)(m0 + 16 + v) * INNER_W + 16 + lm] = acc11[v];
    }
}

// ---------------------------------------------------------------------------
// Host-side launch
// ---------------------------------------------------------------------------
extern "C" void kernel_launch(void* const* d_in, const int* in_sizes, int n_in,
                              void* d_out, int out_size, void* d_ws, size_t ws_size,
                              hipStream_t stream) {
    const float* x   = (const float*)d_in[0];   // (65536, 64)
    const float* ray = (const float*)d_in[1];   // (1, 64)
    const float* T   = (const float*)d_in[2];   // (256, 64, 32)
    const float* w_i = (const float*)d_in[3];   // (1, 255)
    const float* b_i = (const float*)d_in[4];   // (1, 255)
    const float* a_i = (const float*)d_in[5];   // (1, 255)
    float* out = (float*)d_out;                 // (65536, 32)

    char* ws = (char*)d_ws;
    _Float16* Tt  = (_Float16*)ws;                               // 1,179,648 B
    float* rayN   = (float*)(ws + (size_t)T_TOTAL_H * 2);        //   256 B
    float* alphaW = (float*)(ws + (size_t)T_TOTAL_H * 2 + 256);  //  1020 B
    float* betaW  = (float*)(ws + (size_t)T_TOTAL_H * 2 + 256 + 1024);

    prep_misc<<<1, 256, 0, stream>>>(ray, w_i, b_i, a_i, rayN, alphaW, betaW);
    prep_T<<<(N_LEAVES * IN_W * INNER_W + 255) / 256, 256, 0, stream>>>(T, Tt);
    radial_tree_wmma<<<BATCH / BLK_ROWS, BLK_THREADS, 0, stream>>>(
        x, Tt, rayN, alphaW, betaW, out);
}
